// SATG_34677565948058
// MI455X (gfx1250) — matrix operand order repaired and checked by reading, not compiled, not run on hardware
//
#include <hip/hip_runtime.h>

// ---------------------------------------------------------------------------
// CDNA5 (gfx1250, wave32) transformer encoder layer forward.
// All GEMMs via v_wmma_f32_16x16x32_bf16, activations staged bf16 in ws.
// ---------------------------------------------------------------------------

typedef __bf16 v16bf __attribute__((ext_vector_type(16)));
typedef __bf16 bfx8  __attribute__((ext_vector_type(8)));
typedef float  v8f   __attribute__((ext_vector_type(8)));

// A operand (16x32 bf16, MxK): lane m (lr) holds row m; half selects K-subchunk.
// elements 0..7  = A[row][k0 + half*16 .. +8]
// elements 8..15 = A[row][k0 + half*16 + 8 .. +8]
__device__ __forceinline__ v16bf load_a_frag(const __bf16* __restrict__ rowptr,
                                             int k0, int half) {
    bfx8 lo = *(const bfx8*)(rowptr + k0 + half * 16);
    bfx8 hi = *(const bfx8*)(rowptr + k0 + half * 16 + 8);
    v16bf a;
#pragma unroll
    for (int i = 0; i < 8; ++i) { a[i] = lo[i]; a[i + 8] = hi[i]; }
    return a;
}

// B operand (32x16 bf16, KxN) with B[k][n] = W[n][k] (weight stored [N,K]):
// lane n (lr) holds column n; needs W[n][k0 + half*16 .. +16] (contiguous).
__device__ __forceinline__ v16bf load_b_frag(const __bf16* __restrict__ rowptr,
                                             int k0, int half) {
    bfx8 lo = *(const bfx8*)(rowptr + k0 + half * 16);
    bfx8 hi = *(const bfx8*)(rowptr + k0 + half * 16 + 8);
    v16bf b;
#pragma unroll
    for (int i = 0; i < 8; ++i) { b[i] = lo[i]; b[i + 8] = hi[i]; }
    return b;
}

__device__ __forceinline__ v8f wmma_bf16(v16bf a, v16bf b, v8f c) {
    return __builtin_amdgcn_wmma_f32_16x16x32_bf16(false, a, false, b,
                                                   (short)0, c, false, false);
}

// ---------------------------------------------------------------------------
// fp32 -> bf16 conversion
// ---------------------------------------------------------------------------
__global__ void __launch_bounds__(256)
cvt_bf16_kernel(const float* __restrict__ in, __bf16* __restrict__ out, int n) {
    int i = blockIdx.x * 256 + threadIdx.x;
    if (i < n) out[i] = (__bf16)in[i];
}

// ---------------------------------------------------------------------------
// Generic C = A[M,K](bf16) @ W[N,K](bf16)^T + bias, per-wave 16x16 tile.
// flags: 1=+posenc  2=relu  4=+resid(f32,[M,N])  8=write f32  16=write bf16
// ---------------------------------------------------------------------------
#define F_PE    1
#define F_RELU  2
#define F_RES   4
#define F_WF32  8
#define F_WBF  16

__global__ void __launch_bounds__(128)
gemm_bf16_kernel(const __bf16* __restrict__ A, const __bf16* __restrict__ W,
                 const float* __restrict__ bias, float* __restrict__ Cf,
                 __bf16* __restrict__ Cbf, const float* __restrict__ resid,
                 int M, int N, int K, int flags, int S) {
    const int wave = threadIdx.x >> 5;
    const int lane = threadIdx.x & 31;
    const int half = lane >> 4;
    const int lr   = lane & 15;
    const int tilesN = N >> 4;
    const int totalTiles = (M >> 4) * tilesN;
    const int tile = blockIdx.x * 4 + wave;   // wave-uniform
    if (tile >= totalTiles) return;
    const int tm = tile / tilesN, tn = tile % tilesN;
    const int row0 = tm << 4, col0 = tn << 4;

    const __bf16* arow = A + (size_t)(row0 + lr) * K;
    const __bf16* wrow = W + (size_t)(col0 + lr) * K;

    v8f acc = {};
    for (int k0 = 0; k0 < K; k0 += 32) {
        v16bf af = load_a_frag(arow, k0, half);
        v16bf bf = load_b_frag(wrow, k0, half);
        acc = wmma_bf16(af, bf, acc);
    }

#pragma unroll
    for (int v = 0; v < 8; ++v) {
        const int m = v + half * 8;
        const int row = row0 + m, col = col0 + lr;
        float val = acc[v] + bias[col];
        if (flags & F_PE) {   // sinusoidal positional encoding, d_model = 256
            int s = row % S;
            int p2 = col & ~1;  // 2p
            float ang = (float)s * __expf(-0.0359778918f * (float)p2);
            val += (col & 1) ? __cosf(ang) : __sinf(ang);
        }
        if (flags & F_RES)  val += resid[(size_t)row * N + col];
        if (flags & F_RELU) val = fmaxf(val, 0.f);
        const size_t idx = (size_t)row * N + col;
        if (flags & F_WF32) Cf[idx]  = val;
        if (flags & F_WBF)  Cbf[idx] = (__bf16)val;
    }
}

// ---------------------------------------------------------------------------
// QKV projection: qkv = X[M,256] @ in_proj_w[768,256]^T + b, scattered into
//   q,k : [B,H,S,128] bf16     v : transposed [B,H,128,S] bf16
// ---------------------------------------------------------------------------
__global__ void __launch_bounds__(128)
qkv_gemm_kernel(const __bf16* __restrict__ A, const __bf16* __restrict__ W,
                const float* __restrict__ bias, __bf16* __restrict__ qb,
                __bf16* __restrict__ kb, __bf16* __restrict__ vt,
                int M, int S) {
    const int N = 768, K = 256;
    const int wave = threadIdx.x >> 5;
    const int lane = threadIdx.x & 31;
    const int half = lane >> 4, lr = lane & 15;
    const int tilesN = N >> 4;
    const int totalTiles = (M >> 4) * tilesN;
    const int tile = blockIdx.x * 4 + wave;
    if (tile >= totalTiles) return;
    const int tm = tile / tilesN, tn = tile % tilesN;
    const int row0 = tm << 4, col0 = tn << 4;

    const __bf16* arow = A + (size_t)(row0 + lr) * K;
    const __bf16* wrow = W + (size_t)(col0 + lr) * K;

    v8f acc = {};
    for (int k0 = 0; k0 < K; k0 += 32)
        acc = wmma_bf16(load_a_frag(arow, k0, half),
                        load_b_frag(wrow, k0, half), acc);

#pragma unroll
    for (int v = 0; v < 8; ++v) {
        const int m = v + half * 8;
        const int row = row0 + m, col = col0 + lr;
        const float val = acc[v] + bias[col];
        const int b  = row / S, s = row % S;
        const int wh = col >> 8;          // 0=q 1=k 2=v
        const int c2 = col & 255;
        const int h  = c2 >> 7, dd = c2 & 127;
        const int bh = b * 2 + h;
        if (wh == 0)      qb[((size_t)bh * S + s) * 128 + dd] = (__bf16)val;
        else if (wh == 1) kb[((size_t)bh * S + s) * 128 + dd] = (__bf16)val;
        else              vt[((size_t)bh * 128 + dd) * S + s] = (__bf16)val;
    }
}

// ---------------------------------------------------------------------------
// Banded attention. One wave per (b,h,16-query tile). Window: j in [i-99, i].
// Key span [i0-112, i0+15] = 8 aligned 16-key tiles. hd = 128.
// scores: Q(16x128) x K^T via 4 chained wmma per key tile; mask+scale -> LDS;
// softmax in LDS; P bf16 -> LDS; ctx = P(16x128) x V via wmma (V transposed).
// ---------------------------------------------------------------------------
__global__ void __launch_bounds__(128)
attn_kernel(const __bf16* __restrict__ qbuf, const __bf16* __restrict__ kbuf,
            const __bf16* __restrict__ vtbuf, __bf16* __restrict__ ctx,
            int S, int total) {
    __shared__ float  sbuf[4][16][128];                     // 32 KB
    __shared__ __align__(16) __bf16 pbuf[4][16][128];       // 16 KB
    const int wave = threadIdx.x >> 5;
    const int lane = threadIdx.x & 31;
    const int half = lane >> 4, lr = lane & 15;
    const int QT = S >> 4;
    const int task = blockIdx.x * 4 + wave;                 // wave-uniform
    const bool active = task < total;

    int b = 0, h = 0, i0 = 0, bh = 0;
    v16bf qa[4];
    if (active) {
        const int qt = task % QT;
        h  = (task / QT) & 1;
        b  = task / (QT * 2);
        i0 = qt << 4;
        bh = b * 2 + h;
        const __bf16* qrow = qbuf + ((size_t)bh * S + i0 + lr) * 128;
#pragma unroll
        for (int st = 0; st < 4; ++st) qa[st] = load_a_frag(qrow, st * 32, half);

        // ---- scores for 8 key tiles ----
        for (int kt = 0; kt < 8; ++kt) {
            const int j0 = i0 - 112 + kt * 16;
            const int j  = j0 + lr;
            const int jc = j < 0 ? 0 : j;                   // clamped load, masked below
            const __bf16* krow = kbuf + ((size_t)bh * S + jc) * 128;
            v8f acc = {};
#pragma unroll
            for (int st = 0; st < 4; ++st)
                acc = wmma_bf16(qa[st], load_b_frag(krow, st * 32, half), acc);
#pragma unroll
            for (int v = 0; v < 8; ++v) {
                const int m = v + half * 8;
                const int i = i0 + m, jj = j0 + lr;
                const bool ok = (jj >= 0) && (jj <= i) && (jj + 99 >= i);
                sbuf[wave][m][kt * 16 + lr] =
                    ok ? acc[v] * 0.0883883476f : -1e30f;   // 1/sqrt(128)
            }
        }
    }
    __syncthreads();

    // ---- softmax over the 128-wide span (lanes 0..15, one row each) ----
    if (active && lane < 16) {
        const int r = lane;
        float mx = -1e30f;
        for (int c = 0; c < 128; ++c) mx = fmaxf(mx, sbuf[wave][r][c]);
        float sum = 0.f;
        for (int c = 0; c < 128; ++c) {
            float e = __expf(sbuf[wave][r][c] - mx);
            sbuf[wave][r][c] = e;
            sum += e;
        }
        const float inv = 1.f / sum;
        for (int c = 0; c < 128; ++c)
            pbuf[wave][r][c] = (__bf16)(sbuf[wave][r][c] * inv);
    }
    __syncthreads();

    // ---- ctx = P @ V  (V stored transposed: [bh, d, S]) ----
    if (active) {
        for (int dt = 0; dt < 8; ++dt) {
            const int d0 = dt << 4;
            const __bf16* vrow = vtbuf + ((size_t)bh * 128 + d0 + lr) * S;
            v8f acc = {};
#pragma unroll
            for (int st = 0; st < 4; ++st) {
                const int kk0 = st * 32;
                // A operand from P in LDS
                bfx8 plo = *(const bfx8*)&pbuf[wave][lr][kk0 + half * 16];
                bfx8 phi = *(const bfx8*)&pbuf[wave][lr][kk0 + half * 16 + 8];
                v16bf af;
#pragma unroll
                for (int i2 = 0; i2 < 8; ++i2) { af[i2] = plo[i2]; af[i2 + 8] = phi[i2]; }
                // B operand: V^T row d0+lr, keys [start, start+16)
                const int start = i0 - 112 + kk0 + half * 16;
                v16bf bf;
                if (start >= 0) {
                    bfx8 lo = *(const bfx8*)(vrow + start);
                    bfx8 hi = *(const bfx8*)(vrow + start + 8);
#pragma unroll
                    for (int i2 = 0; i2 < 8; ++i2) { bf[i2] = lo[i2]; bf[i2 + 8] = hi[i2]; }
                } else {  // partial/negative span: P is zero there, clamp loads
#pragma unroll
                    for (int i2 = 0; i2 < 16; ++i2) {
                        int jj = start + i2;
                        bf[i2] = vrow[jj < 0 ? 0 : jj];
                    }
                }
                acc = wmma_bf16(af, bf, acc);
            }
#pragma unroll
            for (int v = 0; v < 8; ++v) {
                const int m = v + half * 8;
                ctx[((size_t)(b * S + i0 + m)) * 256 + h * 128 + d0 + lr] =
                    (__bf16)acc[v];
            }
        }
    }
}

// ---------------------------------------------------------------------------
// In-place LayerNorm over D=256 with bf16 mirror. One 256-thread block / row.
// ---------------------------------------------------------------------------
__global__ void __launch_bounds__(256)
ln_kernel(float* __restrict__ x, const float* __restrict__ w,
          const float* __restrict__ bvec, __bf16* __restrict__ xbf) {
    __shared__ float red[256];
    const int row = blockIdx.x, c = threadIdx.x;
    const size_t idx = (size_t)row * 256 + c;
    const float val = x[idx];
    red[c] = val;
    __syncthreads();
    for (int s = 128; s > 0; s >>= 1) {
        if (c < s) red[c] += red[c + s];
        __syncthreads();
    }
    const float mu = red[0] * (1.f / 256.f);
    __syncthreads();
    const float d = val - mu;
    red[c] = d * d;
    __syncthreads();
    for (int s = 128; s > 0; s >>= 1) {
        if (c < s) red[c] += red[c + s];
        __syncthreads();
    }
    const float var = red[0] * (1.f / 256.f);
    const float o = d * rsqrtf(var + 1e-5f) * w[c] + bvec[c];
    x[idx]   = o;
    xbf[idx] = (__bf16)o;
}

// ---------------------------------------------------------------------------
// Output head fc2: logits[row] = dot(u[row, 0:64], fc2_w) + fc2_b
// ---------------------------------------------------------------------------
__global__ void __launch_bounds__(256)
head_kernel(const float* __restrict__ u, const float* __restrict__ w2,
            const float* __restrict__ b2, float* __restrict__ out, int M) {
    int i = blockIdx.x * 256 + threadIdx.x;
    if (i >= M) return;
    float s = 0.f;
#pragma unroll
    for (int k = 0; k < 64; ++k) s += u[(size_t)i * 64 + k] * w2[k];
    out[i] = s + b2[0];
}

// ---------------------------------------------------------------------------
extern "C" void kernel_launch(void* const* d_in, const int* in_sizes, int n_in,
                              void* d_out, int out_size, void* d_ws, size_t ws_size,
                              hipStream_t stream) {
    const float* src  = (const float*)d_in[0];
    // d_in[1] = input_lengths (int64) — unused by the reference math
    const float* Wenc = (const float*)d_in[2];
    const float* benc = (const float*)d_in[3];
    const float* inpw = (const float*)d_in[4];
    const float* inpb = (const float*)d_in[5];
    const float* outw = (const float*)d_in[6];
    const float* outb = (const float*)d_in[7];
    const float* ln1w = (const float*)d_in[8];
    const float* ln1b = (const float*)d_in[9];
    const float* l1w  = (const float*)d_in[10];
    const float* l1b  = (const float*)d_in[11];
    const float* l2w  = (const float*)d_in[12];
    const float* l2b  = (const float*)d_in[13];
    const float* ln2w = (const float*)d_in[14];
    const float* ln2b = (const float*)d_in[15];
    const float* f1w  = (const float*)d_in[16];
    const float* f1b  = (const float*)d_in[17];
    const float* f2w  = (const float*)d_in[18];
    const float* f2b  = (const float*)d_in[19];

    const int Bb = in_sizes[1];                 // batch (8)
    const int S  = in_sizes[0] / (Bb * 256);    // seq len (2000, multiple of 16)
    const int M  = Bb * S;                      // 16000 rows

    // -------- workspace layout (with aliasing reuse) --------
    char* p = (char*)d_ws;
    auto alloc = [&](size_t bytes) -> char* {
        char* r = p;
        p += (bytes + 255) & ~(size_t)255;
        return r;
    };
    __bf16* srcbf = (__bf16*)alloc((size_t)M * 256 * 2);
    __bf16* xbf   = (__bf16*)alloc((size_t)M * 256 * 2);
    float*  xf32  = (float*) alloc((size_t)M * 256 * 4);
    __bf16* qb    = (__bf16*)alloc((size_t)M * 256 * 2);
    __bf16* kb    = (__bf16*)alloc((size_t)M * 256 * 2);
    __bf16* vt    = (__bf16*)alloc((size_t)M * 256 * 2);
    __bf16* wencb = (__bf16*)alloc(65536 * 2);
    __bf16* inpwb = (__bf16*)alloc(196608 * 2);
    __bf16* outwb = (__bf16*)alloc(65536 * 2);
    __bf16* l1wb  = (__bf16*)alloc(65536 * 2);
    __bf16* l2wb  = (__bf16*)alloc(65536 * 2);
    __bf16* f1wb  = (__bf16*)alloc(16384 * 2);
    __bf16* ctx = srcbf;      // src bf16 no longer needed after encoder GEMM
    __bf16* ffb = qb;         // q no longer needed after attention
    float*  ub  = (float*)kb; // k no longer needed after attention

    auto cvt = [&](const float* in, __bf16* out, int n) {
        cvt_bf16_kernel<<<(n + 255) / 256, 256, 0, stream>>>(in, out, n);
    };
    auto gemm = [&](const __bf16* A, const __bf16* W, const float* bias,
                    float* Cf, __bf16* Cbf, const float* resid,
                    int mm, int nn, int flags) {
        const int tiles = (mm >> 4) * (nn >> 4);
        gemm_bf16_kernel<<<(tiles + 3) / 4, 128, 0, stream>>>(
            A, W, bias, Cf, Cbf, resid, mm, nn, 256, flags, S);
    };

    // 1) downconvert weights + src to bf16
    cvt(Wenc, wencb, 65536);
    cvt(inpw, inpwb, 196608);
    cvt(outw, outwb, 65536);
    cvt(l1w,  l1wb,  65536);
    cvt(l2w,  l2wb,  65536);
    cvt(f1w,  f1wb,  16384);
    cvt(src,  srcbf, M * 256);

    // 2) x = src @ Wenc^T + b + posenc        -> xf32, xbf
    gemm(srcbf, wencb, benc, xf32, xbf, nullptr, M, 256,
         F_PE | F_WF32 | F_WBF);

    // 3) qkv = x @ in_proj_w^T + b  -> q,k ([B,H,S,128]) and v^T ([B,H,128,S])
    {
        const int tiles = (M >> 4) * (768 >> 4);
        qkv_gemm_kernel<<<(tiles + 3) / 4, 128, 0, stream>>>(
            xbf, inpwb, inpb, qb, kb, vt, M, S);
    }

    // 4) banded attention -> ctx bf16 [M,256]
    {
        const int total = Bb * 2 * (S >> 4);
        attn_kernel<<<(total + 3) / 4, 128, 0, stream>>>(qb, kb, vt, ctx, S, total);
    }

    // 5) h = x + ctx @ out_proj^T + b   (in-place accumulate into xf32)
    gemm(ctx, outwb, outb, xf32, nullptr, xf32, M, 256, F_RES | F_WF32);

    // 6) layernorm1 (in place) -> xf32, xbf
    ln_kernel<<<M, 256, 0, stream>>>(xf32, ln1w, ln1b, xbf);

    // 7) ff = relu(x @ lin1^T + b)     -> bf16 ffb
    gemm(xbf, l1wb, l1b, nullptr, ffb, nullptr, M, 256, F_RELU | F_WBF);

    // 8) h = x + ff @ lin2^T + b       (in-place into xf32)
    gemm(ffb, l2wb, l2b, xf32, nullptr, xf32, M, 256, F_RES | F_WF32);

    // 9) layernorm2 (in place) -> xf32, xbf
    ln_kernel<<<M, 256, 0, stream>>>(xf32, ln2w, ln2b, xbf);

    // 10) u = x @ fc1^T + b            -> f32 [M,64]
    gemm(xbf, f1wb, f1b, ub, nullptr, nullptr, M, 64, F_WF32);

    // 11) logits = u @ fc2^T + b       -> d_out [B,S] f32
    head_kernel<<<(M + 255) / 256, 256, 0, stream>>>(
        ub, f2w, f2b, (float*)d_out, M);
}
